// QLSTM_68367289418126
// MI455X (gfx1250) — compile-verified, matrix operand order checked by reading, not run on hardware
//
#include <hip/hip_runtime.h>

// QLSTM: B=256, T=512, INPUT=HIDDEN=128, NQ=4.
// Quantum circuit collapsed analytically: RX gates compose (RX(y)RX(th)=RX(y+th)),
// the ring CNOTs are a classical basis permutation, and the product-state
// distribution factorizes, so with Z_q = cos(y_q + theta[g][q]):
//   m[g] = { Z1*Z2*Z3, Z0*Z1, Z0*Z1*Z2, Z0*Z1*Z2*Z3 }
//
// 16 workgroups (16 batch rows each) x 256 threads (8 waves), full T=512
// recurrence in-kernel (batches are independent -> no grid sync).
//   y   = [h,x] @ W_in^T : v_wmma_f32_16x16x32_f16, 1 K-chunk per wave (K=256)
//   m   : cos-products on 64 threads
//   proj: v_wmma_f32_16x16x4_f32 (K=4 exact, f32), 1 col-tile + 4 layers/wave
//   LSTM pointwise with native v_exp/v_rcp/v_tanh; c in regs, h -> LDS f16.
// x_t staging is double-buffered (loads issued a step ahead): 3 barriers/step.

#define B_      256
#define T_      512
#define INPUT_  128
#define HIDDEN_ 128
#define KDIM    256

typedef __attribute__((ext_vector_type(16))) _Float16 v16h;
typedef __attribute__((ext_vector_type(8)))  _Float16 v8h;
typedef __attribute__((ext_vector_type(8)))  float    v8f;
typedef __attribute__((ext_vector_type(4)))  float    v4f;
typedef __attribute__((ext_vector_type(2)))  float    v2f;

#define XH_STRIDE 136   // halves/row: 128+8 pad (272B = 17*16B: banks spread, rows 16B-aligned)
#define YP_STRIDE 36    // floats/row: 8 waves * 4 q + 4 pad (144B = 9*16B)
#define M2_STRIDE 20    // floats/row: 4 g * 4 q + 4 pad (80B = 5*16B)
#define SHUF16 0,1,2,3,4,5,6,7,8,9,10,11,12,13,14,15

#if __has_builtin(__builtin_amdgcn_tanhf)
  #define FAST_TANH(x) __builtin_amdgcn_tanhf(x)
#elif __has_builtin(__builtin_amdgcn_tanh_f32)
  #define FAST_TANH(x) __builtin_amdgcn_tanh_f32(x)
#else
__device__ __forceinline__ float FAST_TANH(float x) {
    // tanh(x) = 1 - 2/(1+exp(2x)); saturates correctly at +-1
    return fmaf(-2.0f, __builtin_amdgcn_rcpf(1.0f + __expf(2.0f * x)), 1.0f);
}
#endif

__device__ __forceinline__ float fast_sigmoid(float x) {
    return __builtin_amdgcn_rcpf(1.0f + __expf(-x));
}

__global__ __launch_bounds__(256, 1)
void qlstm_kernel(const float* __restrict__ x,
                  const float* __restrict__ W_in,
                  const float* __restrict__ b_in,
                  const float* __restrict__ W_out,
                  const float* __restrict__ b_out,
                  const float* __restrict__ thetas,
                  float* __restrict__ out)
{
    __shared__ __align__(16) _Float16 hTile[16 * XH_STRIDE];      // h state, f16
    __shared__ __align__(16) _Float16 xbuf[2][16 * XH_STRIDE];    // x_t double buffer, f16
    __shared__ __align__(16) float    ypart[16 * YP_STRIDE];      // y partials [b][wave][q]
    __shared__ __align__(16) float    m2[16 * M2_STRIDE];         // m [b][g][q]
    __shared__ float yb[4];                                       // b_in
    __shared__ float th[4][4];                                    // thetas[g][q]

    const int tid  = threadIdx.x;
    const int wave = tid >> 5;     // 0..7
    const int lane = tid & 31;
    const int ml   = lane & 15;    // M row (A/C) or N col (B/C)
    const int hi   = lane >> 4;    // half-wave selector
    const int b0   = blockIdx.x * 16;

    // ---- one-time staging ----
    for (int i = tid; i < 16 * XH_STRIDE; i += 256) hTile[i] = (_Float16)0.0f;
    if (tid < 4)  yb[tid] = b_in[tid];
    if (tid < 16) th[tid >> 2][tid & 3] = thetas[tid];

    // Loop-invariant y-GEMM B operand: W_in^T K-chunk `wave` (f16, cols q>=4 zeroed).
    // B 32x16 f16 layout: lane<16 -> K = {0..7,16..23}, lane>=16 -> K = {8..15,24..31}.
    v16h yB;
    {
        const float* gb = W_in + (ml & 3) * KDIM + wave * 32;
        v16h tmp;
        #pragma unroll
        for (int j = 0; j < 8; ++j) {
            tmp[j]     = (_Float16)gb[hi * 8 + j];
            tmp[8 + j] = (_Float16)gb[16 + hi * 8 + j];
        }
        _Float16 msk = (ml < 4) ? (_Float16)1.0f : (_Float16)0.0f;
        yB = tmp * msk;
    }

    // Loop-invariant proj B operand + bias (wave owns col tile ct = wave).
    // B 4x16 f32 layout: lane<16 -> VGPR0=B[0][n], VGPR1=B[1][n]; lane>=16 -> K=2,3.
    v2f wB;
    v8f boutC;
    float cst[8];
    {
        int h = wave * 16 + ml;
        wB.x = W_out[h * 4 + hi * 2 + 0];
        wB.y = W_out[h * 4 + hi * 2 + 1];
        float bo = b_out[h];
        boutC = (v8f){bo, bo, bo, bo, bo, bo, bo, bo};
        #pragma unroll
        for (int r = 0; r < 8; ++r) cst[r] = 0.0f;
    }

    const int xrow = tid >> 4;   // 0..15
    const int xseg = tid & 15;   // 16 segs of 8 floats

    // stage x_0
    {
        const v4f* xp = (const v4f*)(x + ((size_t)(b0 + xrow) * T_) * INPUT_ + xseg * 8);
        v4f a0 = xp[0], a1 = xp[1];
        v8h hx;
        #pragma unroll
        for (int j = 0; j < 4; ++j) { hx[j] = (_Float16)a0[j]; hx[4 + j] = (_Float16)a1[j]; }
        *(v8h*)&xbuf[0][xrow * XH_STRIDE + xseg * 8] = hx;
    }
    __syncthreads();

    const size_t HS = (size_t)B_ * T_ * HIDDEN_;

    for (int t = 0; t < T_; ++t) {
        // ---- issue x_{t+1} global loads early (consumed at end of step) ----
        v4f xr0 = {0.f, 0.f, 0.f, 0.f}, xr1 = {0.f, 0.f, 0.f, 0.f};
        if (t + 1 < T_) {
            const v4f* xp = (const v4f*)(x + ((size_t)(b0 + xrow) * T_ + (t + 1)) * INPUT_ + xseg * 8);
            xr0 = xp[0];
            xr1 = xp[1];
        }

        // ---- Phase B: y partial, one K-chunk per wave (f16 WMMA) ----
        {
            const _Float16* xcur = xbuf[t & 1];
            const _Float16* arow = (wave < 4)
                ? &hTile[ml * XH_STRIDE + wave * 32]
                : &xcur[ml * XH_STRIDE + wave * 32 - 128];
            v8h alo = *(const v8h*)(arow + hi * 8);        // K = base+hi*8 .. +7
            v8h ahi = *(const v8h*)(arow + 16 + hi * 8);   // K = base+16+hi*8 .. +7
            v16h A = __builtin_shufflevector(alo, ahi, SHUF16);
            v8f z = {0.f, 0.f, 0.f, 0.f, 0.f, 0.f, 0.f, 0.f};
            v8f acc = __builtin_amdgcn_wmma_f32_16x16x32_f16(
                          false, A, false, yB, (short)0, z, false, false);
            if (ml < 4) {                                  // only q columns 0..3 meaningful
                #pragma unroll
                for (int r = 0; r < 8; ++r)                // C: lane<16 M=r, lane>=16 M=r+8
                    ypart[(hi * 8 + r) * YP_STRIDE + wave * 4 + ml] = acc[r];
            }
        }
        __syncthreads();

        // ---- Phase C: collapsed circuits; thread (b,g) for tid<64 ----
        if (tid < 64) {
            int b = tid >> 2, g = tid & 3;
            v4f s = {0.f, 0.f, 0.f, 0.f};
            #pragma unroll
            for (int w = 0; w < 8; ++w)
                s += *(const v4f*)&ypart[b * YP_STRIDE + w * 4];
            float Z0 = __cosf(s.x + yb[0] + th[g][0]);
            float Z1 = __cosf(s.y + yb[1] + th[g][1]);
            float Z2 = __cosf(s.z + yb[2] + th[g][2]);
            float Z3 = __cosf(s.w + yb[3] + th[g][3]);
            float p1 = Z0 * Z1, p2 = p1 * Z2;
            v4f mv = {Z1 * Z2 * Z3, p1, p2, p2 * Z3};      // m[g][q], q=0..3
            *(v4f*)&m2[b * M2_STRIDE + g * 4] = mv;        // one ds_store_b128
        }
        __syncthreads();

        // ---- Phase D: proj (f32 WMMA K=4) + LSTM pointwise ----
        {
            v8f D[4];
            #pragma unroll
            for (int g = 0; g < 4; ++g) {
                // A 16x4 f32: lane<16 -> K=0,1; lane>=16 -> K=2,3.
                // [b][g][q] layout => one 8B ds_load_b64 lands the pair directly.
                v2f A2 = *(const v2f*)&m2[ml * M2_STRIDE + g * 4 + hi * 2];
                D[g] = __builtin_amdgcn_wmma_f32_16x16x4_f32(
                           false, A2, false, wB, (short)0, boutC, false, false);
            }
            const int hcol = wave * 16 + ml;
            #pragma unroll
            for (int r = 0; r < 8; ++r) {
                float fv = fast_sigmoid(D[0][r]);
                float iv = fast_sigmoid(D[1][r]);
                float gv = FAST_TANH(D[2][r]);
                float ov = fast_sigmoid(D[3][r]);
                float c  = fv * cst[r] + iv * gv;
                cst[r] = c;
                float hv = ov * FAST_TANH(c);
                int brow = hi * 8 + r;
                hTile[brow * XH_STRIDE + hcol] = (_Float16)hv;
                out[((size_t)(b0 + brow) * T_ + t) * HIDDEN_ + hcol] = hv;
                if (t == T_ - 1) {
                    out[HS + (size_t)(b0 + brow) * HIDDEN_ + hcol] = hv;
                    out[HS + (size_t)B_ * HIDDEN_ + (size_t)(b0 + brow) * HIDDEN_ + hcol] = c;
                }
            }
        }

        // ---- stage x_{t+1} into alternate buffer ----
        if (t + 1 < T_) {
            v8h hx;
            #pragma unroll
            for (int j = 0; j < 4; ++j) { hx[j] = (_Float16)xr0[j]; hx[4 + j] = (_Float16)xr1[j]; }
            *(v8h*)&xbuf[(t + 1) & 1][xrow * XH_STRIDE + xseg * 8] = hx;
        }
        __syncthreads();
    }
}

extern "C" void kernel_launch(void* const* d_in, const int* in_sizes, int n_in,
                              void* d_out, int out_size, void* d_ws, size_t ws_size,
                              hipStream_t stream) {
    (void)in_sizes; (void)n_in; (void)d_ws; (void)ws_size; (void)out_size;
    const float* x      = (const float*)d_in[0];
    const float* W_in   = (const float*)d_in[1];
    const float* b_in   = (const float*)d_in[2];
    const float* W_out  = (const float*)d_in[3];
    const float* b_out  = (const float*)d_in[4];
    const float* thetas = (const float*)d_in[5];
    float* out = (float*)d_out;
    hipLaunchKernelGGL(qlstm_kernel, dim3(16), dim3(256), 0, stream,
                       x, W_in, b_in, W_out, b_out, thetas, out);
}
